// S2UTTransformerDecoder_75823352643901
// MI455X (gfx1250) — matrix operand order repaired and checked by reading, not compiled, not run on hardware
//
#include <hip/hip_runtime.h>

#define DEV_INLINE __device__ __forceinline__

typedef __attribute__((ext_vector_type(16))) __bf16 v16bf;
typedef __attribute__((ext_vector_type(8)))  __bf16 v8bf;
typedef __attribute__((ext_vector_type(8)))  float  v8f;

// Problem constants
constexpr int Bc = 16, Tc = 512, Dc = 512, Hc = 8, DHc = 64, FFc = 2048, OUTc = 1000, Lc = 6;
constexpr int Mrows = Bc * Tc;           // 8192 rows for every GEMM
constexpr float NEGF = -1.0e9f;

DEV_INLINE v16bf combine16(v8bf lo, v8bf hi) {
  return __builtin_shufflevector(lo, hi, 0,1,2,3,4,5,6,7,8,9,10,11,12,13,14,15);
}
// A-fragment (16x32 bf16): lane L, elems 0..15 -> K = 16*(L>>4)+j (contiguous)
DEV_INLINE v16bf load_afrag(const __bf16* p) {
  v8bf lo = *(const v8bf*)p;
  v8bf hi = *(const v8bf*)(p + 8);
  return combine16(lo, hi);
}
DEV_INLINE v8f wmma_bf16(v16bf a, v16bf b, v8f c) {
  return __builtin_amdgcn_wmma_f32_16x16x32_bf16(false, a, false, b, (short)0, c, false, false);
}

// ---------------------------------------------------------------------------
// Weight packing: f32 row-major W[K x N] -> bf16 B-fragment-major.
// Fragment (ntile, kt): lane L holds col = ntile*16 + (L&15),
//   k = kt*32 + 16*(L>>4) + j (j = 0..15, contiguous).  One 32B load per lane.
// ---------------------------------------------------------------------------
__global__ __launch_bounds__(256) void pack_w_kernel(const float* __restrict__ W,
                                                     __bf16* __restrict__ dst,
                                                     int K, int N, int Kt, int Nt) {
  int t = blockIdx.x * 256 + threadIdx.x;
  int frag = t >> 5;
  if (frag >= Kt * Nt) return;
  int L = t & 31;
  int kt = frag % Kt;
  int ntile = frag / Kt;
  int col = ntile * 16 + (L & 15);
  int kb = kt * 32 + ((L >> 4) << 4);
  __bf16* out = dst + (size_t)frag * 512 + (size_t)L * 16;
#pragma unroll
  for (int j = 0; j < 16; ++j) {
    int k = kb + j;
    float v = (k < K && col < N) ? W[(size_t)k * N + col] : 0.f;
    out[j] = (__bf16)v;
  }
}

__global__ __launch_bounds__(256) void cvt_kernel(const float* __restrict__ in,
                                                  __bf16* __restrict__ out, int n) {
  int i = blockIdx.x * 256 + threadIdx.x;
  if (i < n) out[i] = (__bf16)in[i];
}

// pad K dim: f32 [M x Ks] -> bf16 [M x Kd], zero beyond Ks
__global__ __launch_bounds__(256) void padcvt_kernel(const float* __restrict__ in,
                                                     __bf16* __restrict__ out,
                                                     int Ks, int Kd, int total) {
  int i = blockIdx.x * 256 + threadIdx.x;
  if (i >= total) return;
  int r = i / Kd, c = i - r * Kd;
  out[i] = (c < Ks) ? (__bf16)in[(size_t)r * Ks + c] : (__bf16)0.f;
}

// x += sinusoidal PE, emit bf16 copy
__global__ __launch_bounds__(256) void pe_cvt_kernel(float* __restrict__ x,
                                                     __bf16* __restrict__ xbf) {
  int idx = blockIdx.x * 256 + threadIdx.x;   // over Mrows*Dc
  int d = idx & 511;
  int t = (idx >> 9) & 511;
  float freq = __expf((float)(d & ~1) * (-9.2103403720f / 512.0f));
  float ang = (float)t * freq;
  float pe = (d & 1) ? __cosf(ang) : __sinf(ang);
  float v = x[idx] + pe;
  x[idx] = v;
  xbf[idx] = (__bf16)v;
}

// ---------------------------------------------------------------------------
// WMMA GEMM: C[M x N] = A(bf16,row-major,stride Kt*32) @ Wpacked + bias
// block = 256 thr (8 waves); block tile 128x128; wave tile 64x32 (4x2 WMMA C tiles)
// Operands are L2-resident (192MB L2 holds all weights+activations) -> no LDS.
// ---------------------------------------------------------------------------
enum { OM_F32 = 0, OM_BF16_RELU = 1, OM_QK = 2, OM_VT = 3 };

template <int MODE>
__global__ __launch_bounds__(256) void gemm_kernel(const __bf16* __restrict__ A,
                                                   const __bf16* __restrict__ Wp,
                                                   const float* __restrict__ bias,
                                                   void* __restrict__ outp,
                                                   int Kt, int Nout, int Nstore) {
  const int lane = threadIdx.x & 31, wave = threadIdx.x >> 5;
  const int hp = lane >> 4, lc = lane & 15;
  const int wm = wave >> 2, wn = wave & 3;
  const int Kpad = Kt << 5;
  const int rowbase = blockIdx.y * 128 + wm * 64;
  const int colbase = blockIdx.x * 128 + wn * 32;
  const int nt0 = colbase >> 4;
  v8f acc[4][2] = {};
  for (int kt = 0; kt < Kt; ++kt) {
    v16bf af[4];
#pragma unroll
    for (int mt = 0; mt < 4; ++mt) {
      const int row = rowbase + mt * 16 + lc;
      af[mt] = load_afrag(A + (size_t)row * Kpad + kt * 32 + hp * 16);
    }
    v16bf bfm[2];
#pragma unroll
    for (int nt = 0; nt < 2; ++nt)
      bfm[nt] = *(const v16bf*)(Wp + ((size_t)(nt0 + nt) * Kt + kt) * 512 + lane * 16);
#pragma unroll
    for (int mt = 0; mt < 4; ++mt)
#pragma unroll
      for (int nt = 0; nt < 2; ++nt)
        acc[mt][nt] = wmma_bf16(af[mt], bfm[nt], acc[mt][nt]);
  }
  float bv[2];
#pragma unroll
  for (int nt = 0; nt < 2; ++nt) {
    int col = colbase + nt * 16 + lc;
    bv[nt] = (col < Nstore) ? bias[col] : 0.f;
  }
#pragma unroll
  for (int mt = 0; mt < 4; ++mt)
#pragma unroll
    for (int nt = 0; nt < 2; ++nt)
#pragma unroll
      for (int r = 0; r < 8; ++r) {
        int row = rowbase + mt * 16 + r + 8 * hp;   // C layout: VGPR r, half hp
        int col = colbase + nt * 16 + lc;
        float v = acc[mt][nt][r] + bv[nt];
        if (MODE == OM_F32) {
          if (col < Nstore) ((float*)outp)[(size_t)row * Nout + col] = v;
        } else if (MODE == OM_BF16_RELU) {
          ((__bf16*)outp)[(size_t)row * Nout + col] = (__bf16)fmaxf(v, 0.f);
        } else if (MODE == OM_QK) {       // scatter to (B,H,T,DH)
          int b = row >> 9, t2 = row & 511, h = col >> 6, dh = col & 63;
          ((__bf16*)outp)[((size_t)(b * Hc + h) * Tc + t2) * DHc + dh] = (__bf16)v;
        } else {                          // OM_VT: scatter to (B,H,DH,T) (V transposed)
          int b = row >> 9, t2 = row & 511, h = col >> 6, dh = col & 63;
          ((__bf16*)outp)[((size_t)(b * Hc + h) * DHc + dh) * Tc + t2] = (__bf16)v;
        }
      }
}

// ---------------------------------------------------------------------------
// Flash attention (online softmax), wave32 WMMA.
// Q,K: (B,H,T,DH) bf16 ; VT: (B,H,DH,T) bf16 ; Out: (B*T, D) bf16
// block = 256 thr = 8 waves; each wave owns 16 query rows; grid (T/128, H, B)
// ---------------------------------------------------------------------------
__global__ __launch_bounds__(256) void attn_kernel(const __bf16* __restrict__ Q,
                                                   const __bf16* __restrict__ K,
                                                   const __bf16* __restrict__ VT,
                                                   const int* __restrict__ maskv,
                                                   __bf16* __restrict__ Out,
                                                   int causal) {
  __shared__ __align__(32) __bf16 plds[8 * 16 * 32];   // per-wave 16x32 P tile
  const int lane = threadIdx.x & 31, wave = threadIdx.x >> 5;
  const int hp = lane >> 4, lc = lane & 15;
  const int b = blockIdx.z, h = blockIdx.y;
  const int q0 = blockIdx.x * 128 + wave * 16;
  const __bf16* Qb = Q + (size_t)(b * Hc + h) * Tc * DHc;
  const __bf16* Kb = K + (size_t)(b * Hc + h) * Tc * DHc;
  const __bf16* Vb = VT + (size_t)(b * Hc + h) * DHc * Tc;
  const int* mk = maskv + b * Tc;
  __bf16* pw = plds + wave * 512;

  v16bf aq0 = load_afrag(Qb + (size_t)(q0 + lc) * DHc + hp * 16);
  v16bf aq1 = load_afrag(Qb + (size_t)(q0 + lc) * DHc + 32 + hp * 16);

  v8f oacc[4] = {};
  float mrow[8], lrow[8];
#pragma unroll
  for (int r = 0; r < 8; ++r) { mrow[r] = -3.0e38f; lrow[r] = 0.f; }

  for (int kc = 0; kc < Tc; kc += 32) {
    if (causal && kc >= q0 + 16) break;     // wave-uniform: keys fully masked
    float p0[8], p1[8];
#pragma unroll
    for (int c = 0; c < 2; ++c) {
      const int key = kc + c * 16 + lc;     // B-frag col = key row of K
      v16bf bk0 = *(const v16bf*)(Kb + (size_t)key * DHc + hp * 16);
      v16bf bk1 = *(const v16bf*)(Kb + (size_t)key * DHc + 32 + hp * 16);
      v8f sacc = {};
      sacc = wmma_bf16(aq0, bk0, sacc);
      sacc = wmma_bf16(aq1, bk1, sacc);
      const float madd = (mk[key] == 0) ? NEGF : 0.f;
#pragma unroll
      for (int r = 0; r < 8; ++r) {
        int qrow = q0 + r + 8 * hp;
        float v = sacc[r] * 0.125f + madd;  // 1/sqrt(64)
        if (causal && key > qrow) v = NEGF;
        if (c == 0) p0[r] = v; else p1[r] = v;
      }
    }
    // online softmax update over the 32 new columns (row stats live in 16-lane halves)
#pragma unroll
    for (int r = 0; r < 8; ++r) {
      float cm = fmaxf(p0[r], p1[r]);
      for (int m = 1; m < 16; m <<= 1) cm = fmaxf(cm, __shfl_xor(cm, m, 32));
      float mn = fmaxf(mrow[r], cm);
      float scale = __expf(mrow[r] - mn);
      float e0 = __expf(p0[r] - mn);
      float e1 = __expf(p1[r] - mn);
      float rs = e0 + e1;
      for (int m = 1; m < 16; m <<= 1) rs += __shfl_xor(rs, m, 32);
      lrow[r] = lrow[r] * scale + rs;
      mrow[r] = mn;
#pragma unroll
      for (int nt = 0; nt < 4; ++nt) oacc[nt][r] = oacc[nt][r] * scale;
      p0[r] = e0; p1[r] = e1;
    }
    // P (C layout) -> LDS -> A fragment
#pragma unroll
    for (int r = 0; r < 8; ++r) {
      pw[(r + 8 * hp) * 32 + lc]      = (__bf16)p0[r];
      pw[(r + 8 * hp) * 32 + 16 + lc] = (__bf16)p1[r];
    }
    asm volatile("s_wait_dscnt 0" ::: "memory");
    v16bf ap = load_afrag(pw + lc * 32 + hp * 16);
#pragma unroll
    for (int nt = 0; nt < 4; ++nt) {
      v16bf bv = *(const v16bf*)(Vb + (size_t)(nt * 16 + lc) * Tc + kc + hp * 16);
      oacc[nt] = wmma_bf16(ap, bv, oacc[nt]);
    }
  }
#pragma unroll
  for (int nt = 0; nt < 4; ++nt)
#pragma unroll
    for (int r = 0; r < 8; ++r) {
      int qrow = q0 + r + 8 * hp;
      float v = oacc[nt][r] / lrow[r];
      Out[((size_t)b * Tc + qrow) * Dc + h * DHc + nt * 16 + lc] = (__bf16)v;
    }
}

// ---------------------------------------------------------------------------
// LayerNorm: xout = LN(xin (+ add)) * g + b ; also emits bf16 copy.  1 row/block.
// ---------------------------------------------------------------------------
__global__ __launch_bounds__(256) void ln_kernel(const float* __restrict__ xin,
                                                 const float* __restrict__ addv,
                                                 const float* __restrict__ g,
                                                 const float* __restrict__ bta,
                                                 float* __restrict__ xout,
                                                 __bf16* __restrict__ xbf) {
  int row = blockIdx.x, t = threadIdx.x;
  const float* xr = xin + (size_t)row * Dc;
  float v0 = xr[t], v1 = xr[t + 256];
  if (addv) { v0 += addv[(size_t)row * Dc + t]; v1 += addv[(size_t)row * Dc + t + 256]; }
  __shared__ float s1[256], s2[256];
  s1[t] = v0 + v1; s2[t] = v0 * v0 + v1 * v1;
  __syncthreads();
  for (int s = 128; s > 0; s >>= 1) {
    if (t < s) { s1[t] += s1[t + s]; s2[t] += s2[t + s]; }
    __syncthreads();
  }
  float mean = s1[0] * (1.f / 512.f);
  float var = s2[0] * (1.f / 512.f) - mean * mean;
  float rstd = rsqrtf(var + 1e-5f);
  float o0 = (v0 - mean) * rstd * g[t] + bta[t];
  float o1 = (v1 - mean) * rstd * g[t + 256] + bta[t + 256];
  xout[(size_t)row * Dc + t] = o0;
  xout[(size_t)row * Dc + t + 256] = o1;
  xbf[(size_t)row * Dc + t] = (__bf16)o0;
  xbf[(size_t)row * Dc + t + 256] = (__bf16)o1;
}

// in-place softmax over rows of OUTc=1000
__global__ __launch_bounds__(256) void softmax_kernel(float* __restrict__ x) {
  int row = blockIdx.x, t = threadIdx.x;
  float* p = x + (size_t)row * OUTc;
  __shared__ float sd[256];
  float vals[4]; float lm = -3.0e38f;
#pragma unroll
  for (int k2 = 0; k2 < 4; ++k2) {
    int i = t + k2 * 256;
    if (i < OUTc) { vals[k2] = p[i]; lm = fmaxf(lm, vals[k2]); } else vals[k2] = -3.0e38f;
  }
  sd[t] = lm; __syncthreads();
  for (int s = 128; s > 0; s >>= 1) { if (t < s) sd[t] = fmaxf(sd[t], sd[t + s]); __syncthreads(); }
  float mx = sd[0]; __syncthreads();
  float ls = 0.f;
#pragma unroll
  for (int k2 = 0; k2 < 4; ++k2) {
    int i = t + k2 * 256;
    if (i < OUTc) { vals[k2] = __expf(vals[k2] - mx); ls += vals[k2]; }
  }
  sd[t] = ls; __syncthreads();
  for (int s = 128; s > 0; s >>= 1) { if (t < s) sd[t] += sd[t + s]; __syncthreads(); }
  float inv = 1.0f / sd[0];
#pragma unroll
  for (int k2 = 0; k2 < 4; ++k2) {
    int i = t + k2 * 256;
    if (i < OUTc) p[i] = vals[k2] * inv;
  }
}

// ---------------------------------------------------------------------------
extern "C" void kernel_launch(void* const* d_in, const int* in_sizes, int n_in,
                              void* d_out, int out_size, void* d_ws, size_t ws_size,
                              hipStream_t stream) {
  (void)in_sizes; (void)n_in; (void)out_size; (void)ws_size;
  const float* enc  = (const float*)d_in[0];
  const float* tgt  = (const float*)d_in[1];
  const int*  emask = (const int*)d_in[2];
  const int*  tmask = (const int*)d_in[3];
  const float* Wemb = (const float*)d_in[4];
  const float* bemb = (const float*)d_in[5];
  const float *Wq  = (const float*)d_in[6],  *bq  = (const float*)d_in[7];
  const float *Wk  = (const float*)d_in[8],  *bk  = (const float*)d_in[9];
  const float *Wv  = (const float*)d_in[10], *bv  = (const float*)d_in[11];
  const float *Wo  = (const float*)d_in[12], *bo  = (const float*)d_in[13];
  const float *Wq2 = (const float*)d_in[14], *bq2 = (const float*)d_in[15];
  const float *Wk2 = (const float*)d_in[16], *bk2 = (const float*)d_in[17];
  const float *Wv2 = (const float*)d_in[18], *bv2 = (const float*)d_in[19];
  const float *Wo2 = (const float*)d_in[20], *bo2 = (const float*)d_in[21];
  const float *W1  = (const float*)d_in[22], *b1  = (const float*)d_in[23];
  const float *W2  = (const float*)d_in[24], *b2  = (const float*)d_in[25];
  const float *g1  = (const float*)d_in[26], *be1 = (const float*)d_in[27];
  const float *g2  = (const float*)d_in[28], *be2 = (const float*)d_in[29];
  const float *g3  = (const float*)d_in[30], *be3 = (const float*)d_in[31];
  const float *gF  = (const float*)d_in[32], *bF  = (const float*)d_in[33];
  const float *Wout = (const float*)d_in[34], *bout = (const float*)d_in[35];

  char* ws = (char*)d_ws;
  size_t off = 0;
  auto take = [&](size_t bytes) -> char* {
    char* p = ws + off;
    off += (bytes + 255) & ~(size_t)255;
    return p;
  };
  const size_t SZ_DD = (size_t)16 * 32 * 512 * 2;    // 512x512 packed
  const size_t SZ_W1 = (size_t)16 * 128 * 512 * 2;   // 512x2048 packed
  const size_t SZ_W2 = (size_t)64 * 32 * 512 * 2;    // 2048x512 packed

  __bf16 *Wqp[Lc], *Wkp[Lc], *Wvp[Lc], *Wop[Lc], *Wq2p[Lc], *Wk2p[Lc], *Wv2p[Lc], *Wo2p[Lc],
         *W1p[Lc], *W2p[Lc];
  for (int i = 0; i < Lc; ++i) {
    Wqp[i]  = (__bf16*)take(SZ_DD);  Wkp[i]  = (__bf16*)take(SZ_DD);
    Wvp[i]  = (__bf16*)take(SZ_DD);  Wop[i]  = (__bf16*)take(SZ_DD);
    Wq2p[i] = (__bf16*)take(SZ_DD);  Wk2p[i] = (__bf16*)take(SZ_DD);
    Wv2p[i] = (__bf16*)take(SZ_DD);  Wo2p[i] = (__bf16*)take(SZ_DD);
    W1p[i]  = (__bf16*)take(SZ_W1);  W2p[i]  = (__bf16*)take(SZ_W2);
  }
  __bf16* WembP = (__bf16*)take((size_t)32 * 32 * 512 * 2);   // K pad 1024
  __bf16* WoutP = (__bf16*)take((size_t)16 * 64 * 512 * 2);   // N pad 1024
  __bf16* aemb  = (__bf16*)take((size_t)Mrows * 1024 * 2);
  __bf16* encbf = (__bf16*)take((size_t)Mrows * Dc * 2);
  float*  xf    = (float*) take((size_t)Mrows * Dc * 4);
  __bf16* xbf   = (__bf16*)take((size_t)Mrows * Dc * 2);
  float*  proj  = (float*) take((size_t)Mrows * Dc * 4);
  __bf16* qb    = (__bf16*)take((size_t)Mrows * Dc * 2);
  __bf16* kb2   = (__bf16*)take((size_t)Mrows * Dc * 2);
  __bf16* vtb   = (__bf16*)take((size_t)Mrows * Dc * 2);
  __bf16* attnb = (__bf16*)take((size_t)Mrows * Dc * 2);
  __bf16* hbf   = (__bf16*)take((size_t)Mrows * FFc * 2);

  auto packW = [&](const float* W, __bf16* dst, int K, int N) {
    int Kt = (K + 31) / 32;
    int Nt = ((N + 127) / 128) * 8;   // pad N tiles to block-tile multiple
    int total = Kt * Nt * 32;
    pack_w_kernel<<<dim3((total + 255) / 256), dim3(256), 0, stream>>>(W, dst, K, N, Kt, Nt);
  };

  // 1) pack all weights (bf16 fragment layout)
  for (int i = 0; i < Lc; ++i) {
    packW(Wq  + (size_t)i * Dc * Dc,  Wqp[i],  Dc, Dc);
    packW(Wk  + (size_t)i * Dc * Dc,  Wkp[i],  Dc, Dc);
    packW(Wv  + (size_t)i * Dc * Dc,  Wvp[i],  Dc, Dc);
    packW(Wo  + (size_t)i * Dc * Dc,  Wop[i],  Dc, Dc);
    packW(Wq2 + (size_t)i * Dc * Dc,  Wq2p[i], Dc, Dc);
    packW(Wk2 + (size_t)i * Dc * Dc,  Wk2p[i], Dc, Dc);
    packW(Wv2 + (size_t)i * Dc * Dc,  Wv2p[i], Dc, Dc);
    packW(Wo2 + (size_t)i * Dc * Dc,  Wo2p[i], Dc, Dc);
    packW(W1  + (size_t)i * Dc * FFc, W1p[i],  Dc, FFc);
    packW(W2  + (size_t)i * FFc * Dc, W2p[i],  FFc, Dc);
  }
  packW(Wemb, WembP, OUTc, Dc);
  packW(Wout, WoutP, Dc, OUTc);

  // 2) activations to bf16
  {
    int n = Mrows * Dc;
    cvt_kernel<<<dim3(n / 256), dim3(256), 0, stream>>>(enc, encbf, n);
    int total = Mrows * 1024;
    padcvt_kernel<<<dim3(total / 256), dim3(256), 0, stream>>>(tgt, aemb, OUTc, 1024, total);
  }

  const dim3 blk(256);
  const dim3 gD(Dc / 128, Mrows / 128);     // N=512
  const dim3 gFF(FFc / 128, Mrows / 128);   // N=2048
  const dim3 gOUT(1024 / 128, Mrows / 128); // N padded to 1024
  const dim3 gAttn(Tc / 128, Hc, Bc);

  // 3) embedding GEMM (+bias) then positional encoding
  gemm_kernel<OM_F32><<<gD, blk, 0, stream>>>(aemb, WembP, bemb, xf, 32, Dc, Dc);
  pe_cvt_kernel<<<dim3(Mrows * Dc / 256), blk, 0, stream>>>(xf, xbf);

  // 4) decoder layers
  for (int i = 0; i < Lc; ++i) {
    // --- self attention ---
    gemm_kernel<OM_QK><<<gD, blk, 0, stream>>>(xbf, Wqp[i], bq + (size_t)i * Dc, qb, 16, Dc, Dc);
    gemm_kernel<OM_QK><<<gD, blk, 0, stream>>>(xbf, Wkp[i], bk + (size_t)i * Dc, kb2, 16, Dc, Dc);
    gemm_kernel<OM_VT><<<gD, blk, 0, stream>>>(xbf, Wvp[i], bv + (size_t)i * Dc, vtb, 16, Dc, Dc);
    attn_kernel<<<gAttn, blk, 0, stream>>>(qb, kb2, vtb, tmask, attnb, 1);
    gemm_kernel<OM_F32><<<gD, blk, 0, stream>>>(attnb, Wop[i], bo + (size_t)i * Dc, proj, 16, Dc, Dc);
    ln_kernel<<<dim3(Mrows), blk, 0, stream>>>(xf, proj, g1 + (size_t)i * Dc, be1 + (size_t)i * Dc, xf, xbf);
    // --- cross attention ---
    gemm_kernel<OM_QK><<<gD, blk, 0, stream>>>(xbf,   Wq2p[i], bq2 + (size_t)i * Dc, qb, 16, Dc, Dc);
    gemm_kernel<OM_QK><<<gD, blk, 0, stream>>>(encbf, Wk2p[i], bk2 + (size_t)i * Dc, kb2, 16, Dc, Dc);
    gemm_kernel<OM_VT><<<gD, blk, 0, stream>>>(encbf, Wv2p[i], bv2 + (size_t)i * Dc, vtb, 16, Dc, Dc);
    attn_kernel<<<gAttn, blk, 0, stream>>>(qb, kb2, vtb, emask, attnb, 0);
    gemm_kernel<OM_F32><<<gD, blk, 0, stream>>>(attnb, Wo2p[i], bo2 + (size_t)i * Dc, proj, 16, Dc, Dc);
    ln_kernel<<<dim3(Mrows), blk, 0, stream>>>(xf, proj, g2 + (size_t)i * Dc, be2 + (size_t)i * Dc, xf, xbf);
    // --- FFN ---
    gemm_kernel<OM_BF16_RELU><<<gFF, blk, 0, stream>>>(xbf, W1p[i], b1 + (size_t)i * FFc, hbf, 16, FFc, FFc);
    gemm_kernel<OM_F32><<<gD, blk, 0, stream>>>(hbf, W2p[i], b2 + (size_t)i * Dc, proj, 64, Dc, Dc);
    ln_kernel<<<dim3(Mrows), blk, 0, stream>>>(xf, proj, g3 + (size_t)i * Dc, be3 + (size_t)i * Dc, xf, xbf);
  }

  // 5) final LN, logits GEMM, softmax (in-place in d_out)
  ln_kernel<<<dim3(Mrows), blk, 0, stream>>>(xf, nullptr, gF, bF, xf, xbf);
  gemm_kernel<OM_F32><<<gOUT, blk, 0, stream>>>(xbf, WoutP, bout, d_out, 16, OUTc, OUTc);
  softmax_kernel<<<dim3(Mrows), blk, 0, stream>>>((float*)d_out);
}
